// SingleModal_27453430956096
// MI455X (gfx1250) — matrix operand order repaired and checked by reading, not compiled, not run on hardware
//
#include <hip/hip_runtime.h>
#include <math.h>

// ---------------------------------------------------------------------------
// Types for CDNA5 WMMA (wave32, 16x16x32 f16 -> f32)
// ---------------------------------------------------------------------------
typedef _Float16 h8  __attribute__((ext_vector_type(8)));
typedef _Float16 h16 __attribute__((ext_vector_type(16)));
typedef float    f8  __attribute__((ext_vector_type(8)));
typedef int      v4i __attribute__((ext_vector_type(4)));

union H16 { h8 half8x2[2]; h16 v; };

// addrspace-qualified 16-byte chunk types for the async-copy builtin
typedef __attribute__((address_space(1))) v4i gchunk;  // global
typedef __attribute__((address_space(3))) v4i lchunk;  // LDS

#if __has_builtin(__builtin_amdgcn_global_load_async_to_lds_b128)
#define HAVE_ASYNC 1
#else
#define HAVE_ASYNC 0
#endif

// 16-byte global -> LDS copy. Async (ASYNCcnt-tracked, no VGPR data) when the
// toolchain exposes the gfx1250 builtin; plain load/store otherwise.
__device__ __forceinline__ void async_cp16(const _Float16* g, _Float16* l) {
#if HAVE_ASYNC
    __builtin_amdgcn_global_load_async_to_lds_b128(
        (gchunk*)g, (lchunk*)l, /*offset=*/0, /*cpol=*/0);
#else
    *(h8*)l = *(const h8*)g;
#endif
}

__device__ __forceinline__ void async_wait0() {
#if HAVE_ASYNC
#if __has_builtin(__builtin_amdgcn_s_wait_asynccnt)
    __builtin_amdgcn_s_wait_asynccnt(0);
#else
    asm volatile("s_wait_asynccnt 0x0" ::: "memory");
#endif
#endif
}

__device__ __forceinline__ f8 f8zero() {
    f8 z;
#pragma unroll
    for (int i = 0; i < 8; ++i) z[i] = 0.0f;
    return z;
}

__device__ __forceinline__ f8 wmma_f16(h16 a, h16 b, f8 c) {
    return __builtin_amdgcn_wmma_f32_16x16x32_f16(
        false, a, false, b, (short)0, c, false, false);
}

// ---------------------------------------------------------------------------
// fp32 -> f16 conversion (weights / activations)
// ---------------------------------------------------------------------------
__global__ void cvt_f32_to_f16(const float* __restrict__ src,
                               _Float16* __restrict__ dst, int n)
{
    const int i = blockIdx.x * 256 + threadIdx.x;
    if (i < n) dst[i] = (_Float16)src[i];
}

// ---------------------------------------------------------------------------
// WMMA GEMM: out[M,N] = act( A[M,K](f16) @ W[N,K](f16)^T + bias[N] ) (+R)
// Block tile 128x64, 8 waves; K%32==0, M%128==0, N%64==0 (true here).
// ACT: 0=none, 1=exact GELU.  RES: add f32 residual.  OUTF16: f16 output.
// ---------------------------------------------------------------------------
#define BK 32

template <int ACT, int RES, int OUTF16>
__global__ void __launch_bounds__(256)
gemm_wmma(const _Float16* __restrict__ Ag, const _Float16* __restrict__ Wg,
          const float* __restrict__ bias, const float* __restrict__ Rg,
          float* __restrict__ Y32, _Float16* __restrict__ Y16,
          int M, int N, int K)
{
    __shared__ _Float16 As[128 * BK];
    __shared__ _Float16 Bs[64 * BK];

    const int tid  = threadIdx.x;
    const int lane = tid & 31;
    const int wave = tid >> 5;
    const int hr   = lane & 15;
    const int hs   = lane >> 4;
    const int m0   = blockIdx.x * 128;
    const int n0   = blockIdx.y * 64;

    f8 acc[4];
#pragma unroll
    for (int t = 0; t < 4; ++t) acc[t] = f8zero();

    const int seg  = tid & 3;   // 8-half (16 B) chunk within a 32-half row
    const int rowc = tid >> 2;  // 0..63

    for (int k0 = 0; k0 < K; k0 += BK) {
        __syncthreads();
        // stage A tile 128x32 f16: 2 chunks/thread
#pragma unroll
        for (int p = 0; p < 2; ++p) {
            const int r = rowc + p * 64;
            async_cp16(Ag + (size_t)(m0 + r) * K + k0 + seg * 8, &As[r * BK + seg * 8]);
        }
        // stage B tile 64x32 f16: 1 chunk/thread
        async_cp16(Wg + (size_t)(n0 + rowc) * K + k0 + seg * 8, &Bs[rowc * BK + seg * 8]);
        async_wait0();
        __syncthreads();

        // A fragment (contiguous k = 16*hs + s, same order as B)
        H16 ua;
        const _Float16* ap = &As[(wave * 16 + hr) * BK + hs * 16];
        ua.half8x2[0] = *(const h8*)(ap);
        ua.half8x2[1] = *(const h8*)(ap + 8);
#pragma unroll
        for (int t = 0; t < 4; ++t) {
            H16 ub;
            const _Float16* bp_ = &Bs[(t * 16 + hr) * BK + hs * 16];
            ub.half8x2[0] = *(const h8*)(bp_);
            ub.half8x2[1] = *(const h8*)(bp_ + 8);
            acc[t] = wmma_f16(ua.v, ub.v, acc[t]);
        }
    }

#pragma unroll
    for (int t = 0; t < 4; ++t) {
        const int nc = n0 + t * 16 + hr;
        const float bv = bias[nc];
#pragma unroll
        for (int r = 0; r < 8; ++r) {
            const int mr = m0 + wave * 16 + hs * 8 + r;
            float v = acc[t][r] + bv;
            if (ACT == 1) v = 0.5f * v * (1.0f + erff(v * 0.70710678118654752f));
            if (RES) v += Rg[(size_t)mr * N + nc];
            if (OUTF16) Y16[(size_t)mr * N + nc] = (_Float16)v;
            else        Y32[(size_t)mr * N + nc] = v;
        }
    }
}

// ---------------------------------------------------------------------------
// LayerNorm (E=512): one wave per row; reads f32 residual stream, emits f16.
// ---------------------------------------------------------------------------
__global__ void __launch_bounds__(256)
layernorm_k(const float* __restrict__ X, const float* __restrict__ g,
            const float* __restrict__ b, _Float16* __restrict__ Y, int nrows)
{
    const int wave = threadIdx.x >> 5, lane = threadIdx.x & 31;
    const int row  = blockIdx.x * 8 + wave;
    if (row >= nrows) return;
    const float* x = X + (size_t)row * 512;
    float s = 0.f, s2 = 0.f, vals[16];
#pragma unroll
    for (int i = 0; i < 16; ++i) {
        float v = x[lane + i * 32];
        vals[i] = v; s += v; s2 += v * v;
    }
#pragma unroll
    for (int off = 16; off >= 1; off >>= 1) {
        s  += __shfl_xor(s,  off, 32);
        s2 += __shfl_xor(s2, off, 32);
    }
    const float mean = s * (1.0f / 512.0f);
    const float var  = s2 * (1.0f / 512.0f) - mean * mean;
    const float rstd = rsqrtf(var + 1e-5f);
    _Float16* y = Y + (size_t)row * 512;
#pragma unroll
    for (int i = 0; i < 16; ++i) {
        const int c = lane + i * 32;
        y[c] = (_Float16)((vals[i] - mean) * rstd * g[c] + b[c]);
    }
}

// ---------------------------------------------------------------------------
// Flash attention (non-causal). B=16,S=1024,H=8,DH=64. QKV is f16 [M,1536].
// Grid (B*H, S/128); 8 waves; each wave: 16 query rows. 1/sqrt(DH) folded
// into the softmax reads so Q/K staging is a pure byte copy (async path).
// ---------------------------------------------------------------------------
__global__ void __launch_bounds__(256)
attn_flash(const _Float16* __restrict__ QKV, _Float16* __restrict__ O)
{
    __shared__ _Float16 Qs[128 * 64];
    __shared__ _Float16 Ks[64 * 64];
    __shared__ _Float16 Vt[64 * 64];     // transposed [d][kcol]
    __shared__ _Float16 Ps[8 * 16 * 64]; // per-wave P tile

    const int tid = threadIdx.x, lane = tid & 31, wave = tid >> 5;
    const int hr = lane & 15, hs = lane >> 4;
    const int bh = blockIdx.x;
    const int b  = bh >> 3, h = bh & 7;
    const int q0 = blockIdx.y * 128;
    const float scale = 0.125f;
    const size_t baseQ = (size_t)b * 1024 * 1536 + (size_t)h * 64;

    const int seg8 = tid & 7;   // 8-half chunk within a 64-half row
    const int rowc = tid >> 3;  // 0..31

    // stage Q tile 128x64 (async copy)
#pragma unroll
    for (int p = 0; p < 4; ++p) {
        const int r = rowc + p * 32;
        async_cp16(QKV + baseQ + (size_t)(q0 + r) * 1536 + seg8 * 8, &Qs[r * 64 + seg8 * 8]);
    }

    f8 accO[4];
#pragma unroll
    for (int t = 0; t < 4; ++t) accO[t] = f8zero();
    float mrow[8], lrow[8];
#pragma unroll
    for (int r = 0; r < 8; ++r) { mrow[r] = -1e30f; lrow[r] = 0.f; }

    for (int kt = 0; kt < 1024; kt += 64) {
        __syncthreads();
        // stage K tile (async) and transposed V tile (manual)
#pragma unroll
        for (int p = 0; p < 2; ++p) {
            const int r = rowc + p * 32;
            async_cp16(QKV + baseQ + 512 + (size_t)(kt + r) * 1536 + seg8 * 8,
                       &Ks[r * 64 + seg8 * 8]);
        }
#pragma unroll
        for (int p = 0; p < 2; ++p) {
            const int kcol = rowc + p * 32;
            const int d0 = seg8 * 8;
            const h8 v = *(const h8*)(QKV + baseQ + 1024 + (size_t)(kt + kcol) * 1536 + d0);
#pragma unroll
            for (int j = 0; j < 8; ++j) Vt[(d0 + j) * 64 + kcol] = v[j];
        }
        async_wait0();
        __syncthreads();

        // S = Q @ K^T
        f8 accS[4];
#pragma unroll
        for (int t = 0; t < 4; ++t) accS[t] = f8zero();
#pragma unroll
        for (int kk = 0; kk < 64; kk += 32) {
            H16 ua;
            const _Float16* ap = &Qs[(wave * 16 + hr) * 64 + kk + hs * 16];
            ua.half8x2[0] = *(const h8*)(ap);
            ua.half8x2[1] = *(const h8*)(ap + 8);
#pragma unroll
            for (int t = 0; t < 4; ++t) {
                H16 ub;
                const _Float16* bp_ = &Ks[(t * 16 + hr) * 64 + kk + hs * 16];
                ub.half8x2[0] = *(const h8*)(bp_);
                ub.half8x2[1] = *(const h8*)(bp_ + 8);
                accS[t] = wmma_f16(ua.v, ub.v, accS[t]);
            }
        }

        // online softmax (scale folded in here)
        _Float16* pw = &Ps[wave * 16 * 64];
#pragma unroll
        for (int r = 0; r < 8; ++r) {
            float mx = fmaxf(fmaxf(accS[0][r], accS[1][r]),
                             fmaxf(accS[2][r], accS[3][r])) * scale;
#pragma unroll
            for (int off = 8; off >= 1; off >>= 1) mx = fmaxf(mx, __shfl_xor(mx, off, 32));
            const float mnew = fmaxf(mrow[r], mx);
            const float corr = __expf(mrow[r] - mnew);
            float psum = 0.f;
#pragma unroll
            for (int t = 0; t < 4; ++t) {
                const float p = __expf(accS[t][r] * scale - mnew);
                psum += p;
                pw[(hs * 8 + r) * 64 + t * 16 + hr] = (_Float16)p;
            }
#pragma unroll
            for (int off = 8; off >= 1; off >>= 1) psum += __shfl_xor(psum, off, 32);
            lrow[r] = lrow[r] * corr + psum;
            mrow[r] = mnew;
#pragma unroll
            for (int t = 0; t < 4; ++t) accO[t][r] *= corr;
        }
        __syncthreads();

        // O += P @ V
#pragma unroll
        for (int kk = 0; kk < 64; kk += 32) {
            H16 ua;
            const _Float16* ap = &pw[hr * 64 + kk + hs * 16];
            ua.half8x2[0] = *(const h8*)(ap);
            ua.half8x2[1] = *(const h8*)(ap + 8);
#pragma unroll
            for (int t = 0; t < 4; ++t) {
                H16 ub;
                const _Float16* bp_ = &Vt[(t * 16 + hr) * 64 + kk + hs * 16];
                ub.half8x2[0] = *(const h8*)(bp_);
                ub.half8x2[1] = *(const h8*)(bp_ + 8);
                accO[t] = wmma_f16(ua.v, ub.v, accO[t]);
            }
        }
    }

    // normalize; write f16 head slice into [B,S,E]
#pragma unroll
    for (int t = 0; t < 4; ++t) {
        const int d = t * 16 + hr;
#pragma unroll
        for (int r = 0; r < 8; ++r) {
            const int q = q0 + wave * 16 + hs * 8 + r;
            O[((size_t)b * 1024 + q) * 512 + h * 64 + d] =
                (_Float16)(accO[t][r] * (1.0f / lrow[r]));
        }
    }
}

// ---------------------------------------------------------------------------
// Mamba-style scan: persistent single workgroup over S=1024 steps.
// comb (f16 LDS) = [x_t | h]; h = max(tanh(comb @ Wi2h^T + b), 0).
// x_t tiles arrive via async global->LDS copies from the f16 copy of X.
// ---------------------------------------------------------------------------
__global__ void __launch_bounds__(256)
mamba_scan(const _Float16* __restrict__ X16,   // [16,1024,512] f16
           const float* __restrict__ h0,       // [16,512]
           const _Float16* __restrict__ Wf16,  // [512,1024]
           const float* __restrict__ bias,     // [512]
           float* __restrict__ hout)           // [16,512]
{
    __shared__ _Float16 comb[16 * 1024];
    const int tid = threadIdx.x, lane = tid & 31, wave = tid >> 5;
    const int hr = lane & 15, hs = lane >> 4;

    for (int i = tid; i < 16 * 512; i += 256) {
        const int bb = i >> 9, e = i & 511;
        comb[bb * 1024 + 512 + e] = (_Float16)h0[i];
    }
    float bvals[4];
#pragma unroll
    for (int t = 0; t < 4; ++t) bvals[t] = bias[wave * 64 + t * 16 + hr];

    for (int s = 0; s < 1024; ++s) {
        __syncthreads();
        // stage x_t: 16x512 f16 = 1024 x 16B chunks, 4 per thread (async)
#pragma unroll
        for (int p = 0; p < 4; ++p) {
            const int chunk = tid + p * 256;
            const int bb = chunk >> 6, c = chunk & 63;
            async_cp16(X16 + ((size_t)bb * 1024 + s) * 512 + c * 8,
                       &comb[bb * 1024 + c * 8]);
        }
        async_wait0();
        __syncthreads();

        f8 acc[4];
#pragma unroll
        for (int t = 0; t < 4; ++t) acc[t] = f8zero();
        for (int kk = 0; kk < 1024; kk += 32) {
            H16 ua;
            const _Float16* ap = &comb[hr * 1024 + kk + hs * 16];
            ua.half8x2[0] = *(const h8*)(ap);
            ua.half8x2[1] = *(const h8*)(ap + 8);
            __builtin_prefetch(Wf16 + (size_t)(wave * 64 + hr) * 1024 + kk + 64, 0, 2);
#pragma unroll
            for (int t = 0; t < 4; ++t) {
                const int n = wave * 64 + t * 16 + hr;
                const _Float16* bp_ = Wf16 + (size_t)n * 1024 + kk + hs * 16;
                H16 ub;
                ub.half8x2[0] = *(const h8*)(bp_);
                ub.half8x2[1] = *(const h8*)(bp_ + 8);
                acc[t] = wmma_f16(ua.v, ub.v, acc[t]);
            }
        }
        __syncthreads();  // all comb reads done before h overwrite
#pragma unroll
        for (int t = 0; t < 4; ++t) {
            const int n = wave * 64 + t * 16 + hr;
#pragma unroll
            for (int r = 0; r < 8; ++r) {
                const int bb = hs * 8 + r;
                float hv = tanhf(acc[t][r] + bvals[t]);
                hv = fmaxf(hv, 0.0f);
                comb[bb * 1024 + 512 + n] = (_Float16)hv;
            }
        }
    }
    __syncthreads();
    for (int i = tid; i < 16 * 512; i += 256) {
        const int bb = i >> 9, e = i & 511;
        hout[i] = (float)comb[bb * 1024 + 512 + e];
    }
}

// ---------------------------------------------------------------------------
// Head kernels
// ---------------------------------------------------------------------------
__global__ void h2o_kernel(const float* __restrict__ hin, const float* __restrict__ W,
                           const float* __restrict__ b, float* __restrict__ out)
{
    const int idx = blockIdx.x * 256 + threadIdx.x;  // 8192 = 16*512
    const int bb = idx >> 9, e = idx & 511;
    const float* hp = hin + bb * 512;
    const float* wp = W + e * 512;
    float s = b[e];
    for (int k = 0; k < 512; ++k) s += hp[k] * wp[k];
    out[idx] = s;
}

__global__ void classifier_kernel(const float* __restrict__ out1, const float* __restrict__ Wc,
                                  const float* __restrict__ bc, float* __restrict__ res)
{
    const int wave = threadIdx.x >> 5, lane = threadIdx.x & 31;  // 16 waves
    float s = 0.f;
    for (int e = lane; e < 512; e += 32) s += out1[wave * 512 + e] * Wc[e];
#pragma unroll
    for (int off = 16; off >= 1; off >>= 1) s += __shfl_xor(s, off, 32);
    if (lane == 0) {
        const float lg = s + bc[0];
        res[wave]      = lg;
        res[16 + wave] = 1.0f / (1.0f + __expf(-lg));
    }
}

// ---------------------------------------------------------------------------
// Orchestration
// ---------------------------------------------------------------------------
extern "C" void kernel_launch(void* const* d_in, const int* in_sizes, int n_in,
                              void* d_out, int out_size, void* d_ws, size_t ws_size,
                              hipStream_t stream)
{
    (void)in_sizes; (void)n_in; (void)out_size; (void)ws_size;

    const float* bm   = (const float*)d_in[0];
    const float* h0   = (const float*)d_in[1];
    const float* Wp   = (const float*)d_in[2];
    const float* bp   = (const float*)d_in[3];
    const float* ln1g = (const float*)d_in[4];
    const float* ln1b = (const float*)d_in[5];
    const float* Wqkv = (const float*)d_in[6];
    const float* bqkv = (const float*)d_in[7];
    const float* Wo   = (const float*)d_in[8];
    const float* bo   = (const float*)d_in[9];
    const float* ln2g = (const float*)d_in[10];
    const float* ln2b = (const float*)d_in[11];
    const float* W1   = (const float*)d_in[12];
    const float* b1   = (const float*)d_in[13];
    const float* W2   = (const float*)d_in[14];
    const float* b2   = (const float*)d_in[15];
    const float* Wi2h = (const float*)d_in[16];
    const float* bi2h = (const float*)d_in[17];
    const float* Wh2o = (const float*)d_in[18];
    const float* bh2o = (const float*)d_in[19];
    const float* Wc   = (const float*)d_in[20];
    const float* bc   = (const float*)d_in[21];

    const int M = 16 * 1024;
    char* ws = (char*)d_ws;
    size_t off = 0;
    auto alloc = [&](size_t bytes) -> void* {
        void* p = ws + off;
        off += (bytes + 255) & ~(size_t)255;
        return p;
    };
    float*    X     = (float*)alloc((size_t)M * 512 * 4);      // residual stream
    _Float16* X16   = (_Float16*)alloc((size_t)M * 512 * 2);   // f16 copy for scan
    _Float16* Hb    = (_Float16*)alloc((size_t)M * 512 * 2);   // LN output
    _Float16* QKV16 = (_Float16*)alloc((size_t)M * 1536 * 2);
    _Float16* AO16  = (_Float16*)alloc((size_t)M * 512 * 2);
    _Float16* FF116 = (_Float16*)alloc((size_t)M * 2048 * 2);
    _Float16* bm16  = (_Float16*)alloc((size_t)M * 768 * 2);
    _Float16* Wp16  = (_Float16*)alloc((size_t)512 * 768 * 2);
    _Float16* Wqkv6 = (_Float16*)alloc((size_t)2 * 1536 * 512 * 2);
    _Float16* Wo16  = (_Float16*)alloc((size_t)2 * 512 * 512 * 2);
    _Float16* W116  = (_Float16*)alloc((size_t)2 * 2048 * 512 * 2);
    _Float16* W216  = (_Float16*)alloc((size_t)2 * 512 * 2048 * 2);
    _Float16* Wi16  = (_Float16*)alloc((size_t)512 * 1024 * 2);
    float*    hfin  = (float*)alloc(16 * 512 * 4);
    float*    out1  = (float*)alloc(16 * 512 * 4);

    const dim3 blk(256);
    auto cvt = [&](const float* s, _Float16* d, int n) {
        cvt_f32_to_f16<<<dim3((n + 255) / 256), blk, 0, stream>>>(s, d, n);
    };

    // one-time f16 conversions (weights + input activations)
    cvt(bm,   bm16,  M * 768);
    cvt(Wp,   Wp16,  512 * 768);
    cvt(Wqkv, Wqkv6, 2 * 1536 * 512);
    cvt(Wo,   Wo16,  2 * 512 * 512);
    cvt(W1,   W116,  2 * 2048 * 512);
    cvt(W2,   W216,  2 * 512 * 2048);
    cvt(Wi2h, Wi16,  512 * 1024);

    // modality projection: X = bm @ Wp^T + bp  (f32 out, residual stream)
    gemm_wmma<0, 0, 0><<<dim3(M / 128, 512 / 64), blk, 0, stream>>>(
        bm16, Wp16, bp, nullptr, X, nullptr, M, 512, 768);

    for (int l = 0; l < 2; ++l) {
        layernorm_k<<<dim3(M / 8), blk, 0, stream>>>(X, ln1g + l * 512, ln1b + l * 512, Hb, M);
        gemm_wmma<0, 0, 1><<<dim3(M / 128, 1536 / 64), blk, 0, stream>>>(
            Hb, Wqkv6 + (size_t)l * 1536 * 512, bqkv + l * 1536, nullptr,
            nullptr, QKV16, M, 1536, 512);
        attn_flash<<<dim3(16 * 8, 1024 / 128), blk, 0, stream>>>(QKV16, AO16);
        gemm_wmma<0, 1, 0><<<dim3(M / 128, 512 / 64), blk, 0, stream>>>(
            AO16, Wo16 + (size_t)l * 512 * 512, bo + l * 512, X, X, nullptr, M, 512, 512);
        layernorm_k<<<dim3(M / 8), blk, 0, stream>>>(X, ln2g + l * 512, ln2b + l * 512, Hb, M);
        gemm_wmma<1, 0, 1><<<dim3(M / 128, 2048 / 64), blk, 0, stream>>>(
            Hb, W116 + (size_t)l * 2048 * 512, b1 + l * 2048, nullptr,
            nullptr, FF116, M, 2048, 512);
        gemm_wmma<0, 1, 0><<<dim3(M / 128, 512 / 64), blk, 0, stream>>>(
            FF116, W216 + (size_t)l * 512 * 2048, b2 + l * 512, X, X, nullptr, M, 512, 2048);
    }

    // Mamba scan over S on an f16 copy of X
    cvt(X, X16, M * 512);
    mamba_scan<<<dim3(1), blk, 0, stream>>>(X16, h0, Wi16, bi2h, hfin);
    h2o_kernel<<<dim3(16 * 512 / 256), blk, 0, stream>>>(hfin, Wh2o, bh2o, out1);
    classifier_kernel<<<dim3(1), dim3(512), 0, stream>>>(out1, Wc, bc, (float*)d_out);
}
